// FusionHead_63848983823057
// MI455X (gfx1250) — compile-verified, matrix operand order checked by reading, not run on hardware
//
#include <hip/hip_runtime.h>
#include <math.h>

// ---------------- constants ----------------
#define Dm     192          // d_model
#define SSTR   200          // bf16 LDS row stride (ushorts) for 192-wide buffers
#define QSTR   584          // bf16 LDS row stride (ushorts) for 576-wide buffer
#define ROWS   32           // batch rows per workgroup
#define TROWS  64           // token rows (2 tokens * 32 batch rows)
#define BLOCK  256          // 8 wave32
#define NWAVES 8

typedef __attribute__((ext_vector_type(16))) __bf16          v16bf;
typedef __attribute__((ext_vector_type(8)))  unsigned short  u16x8;
typedef __attribute__((ext_vector_type(8)))  float           v8f;

static __device__ __forceinline__ unsigned short f2bf(float f) {
    return (unsigned short)(__float_as_uint(f) >> 16);
}
static __device__ __forceinline__ float bf2f(unsigned short u) {
    return __uint_as_float(((unsigned int)u) << 16);
}

struct FragBits { u16x8 lo, hi; };   // 32 bytes, matches v16bf

// One 16x32 bf16 fragment: lanes hold K = kA..kA+7 (lo) and kA+16..kA+23 (hi),
// both 16-byte aligned -> two b128 loads, no conversion VALU.
static __device__ __forceinline__ v16bf load_frag(const unsigned short* base) {
    const u16x8* p = (const u16x8*)base;
    FragBits f;
    f.lo = p[0];
    f.hi = p[2];                      // +16 ushorts
    return __builtin_bit_cast(v16bf, f);
}

// WMMA GEMM over a tile grid: Out[M x N] = act(A[M x K] * W[N x K]^T + bias)
// A: bf16 in LDS. W: bf16 prepacked in global (row n K-contiguous). Out: bf16 LDS.
static __device__ __forceinline__ void mm_tiles(
    const unsigned short* __restrict__ A, int aStride,
    const unsigned short* __restrict__ W, const float* __restrict__ bias,
    unsigned short* __restrict__ Out, int oStride, bool relu,
    int Mtiles, int Ntiles, int K, int wave, int lane)
{
    const int col  = lane & 15;
    const int half = lane >> 4;
    const int tiles = Mtiles * Ntiles;
    for (int t = wave; t < tiles; t += NWAVES) {
        const int mt = t / Ntiles;
        const int nt = t - mt * Ntiles;
        const int arow = mt * 16 + col;   // A layout: lanes 0-15 & 16-31 both carry M = col
        const int ncol = nt * 16 + col;   // B layout: column n = col

        // prefetch next tile's weight row (gfx1250 global_prefetch_b8)
        const int t2 = t + NWAVES;
        if (t2 < tiles) {
            const int nc2 = (t2 - (t2 / Ntiles) * Ntiles) * 16 + col;
            __builtin_prefetch(W + (size_t)nc2 * K, 0, 0);
        }

        v8f acc = {};
        const unsigned short* Arow = A + (size_t)arow * aStride;
        const unsigned short* Wrow = W + (size_t)ncol * K;
        #pragma unroll
        for (int k0 = 0; k0 < K; k0 += 32) {
            const int kA = k0 + half * 8;
            acc = __builtin_amdgcn_wmma_f32_16x16x32_bf16(
                false, load_frag(Arow + kA),
                false, load_frag(Wrow + kA),
                (short)0, acc, false, false);
        }
        const float bv = bias[nt * 16 + col];
        #pragma unroll
        for (int r = 0; r < 8; ++r) {
            const int m = mt * 16 + half * 8 + r;   // C/D: VGPR r -> M = r (+8 hi lanes)
            float v = acc[r] + bv;
            if (relu) v = fmaxf(v, 0.0f);
            Out[(size_t)m * oStride + nt * 16 + col] = f2bf(v);
        }
    }
}

// x = LayerNorm(x + r) * g + b over TROWS bf16 rows of width Dm (fp32 math).
static __device__ __forceinline__ void ln_residual(
    unsigned short* __restrict__ x, const unsigned short* __restrict__ r,
    const float* __restrict__ g, const float* __restrict__ bb, int tid)
{
    for (int row = tid; row < TROWS; row += BLOCK) {
        float s = 0.f, q = 0.f;
        for (int c = 0; c < Dm; ++c) {
            const float v = bf2f(x[row * SSTR + c]) + bf2f(r[row * SSTR + c]);
            s += v; q += v * v;
        }
        const float mu  = s * (1.0f / Dm);
        const float var = q * (1.0f / Dm) - mu * mu;
        const float rs  = rsqrtf(var + 1e-5f);
        for (int c = 0; c < Dm; ++c) {
            const float v = bf2f(x[row * SSTR + c]) + bf2f(r[row * SSTR + c]);
            x[row * SSTR + c] = f2bf((v - mu) * rs * g[c] + bb[c]);
        }
    }
}

// bf16 weight-prepack offsets inside d_ws (ushort units)
#define OFF_WP   0
#define OFF_WT   (OFF_WP + Dm * Dm)
#define OFF_IN   (OFF_WT + Dm * Dm)
#define OFF_OUT  (OFF_IN + 3 * Dm * Dm)
#define OFF_F1   (OFF_OUT + Dm * Dm)
#define OFF_F2   (OFF_F1 + 2 * Dm * Dm)
#define OFF_CLS1 (OFF_F2 + 2 * Dm * Dm)
#define OFF_FP   (OFF_CLS1 + Dm * Dm)
#define WS_ELEMS (OFF_FP + 128 * Dm)

__global__ void cvt_bf16_kernel(const float* __restrict__ src,
                                unsigned short* __restrict__ dst, int n) {
    const int i = blockIdx.x * blockDim.x + threadIdx.x;
    if (i < n) dst[i] = f2bf(src[i]);
}

__global__ void __launch_bounds__(BLOCK)
fusion_head_kernel(
    const float* __restrict__ perc, const float* __restrict__ tech,
    const unsigned short* __restrict__ wsw,   // prepacked bf16 weights
    const float* __restrict__ bp,  const float* __restrict__ bt,
    const float* __restrict__ in_b, const float* __restrict__ out_b,
    const float* __restrict__ ffn_b1, const float* __restrict__ ffn_b2,
    const float* __restrict__ ln1_g, const float* __restrict__ ln1_b,
    const float* __restrict__ ln2_g, const float* __restrict__ ln2_b,
    const float* __restrict__ cls_b1,
    const float* __restrict__ cls_w2, const float* __restrict__ cls_b2,
    const float* __restrict__ fp_b,
    float* __restrict__ out, int B)
{
    extern __shared__ char smem[];
    unsigned short* sSeq = (unsigned short*)smem;          // [TROWS][SSTR]
    unsigned short* sTmp = sSeq + TROWS * SSTR;            // [TROWS][SSTR]
    unsigned short* sQ   = sTmp + TROWS * SSTR;            // [TROWS][QSTR]

    const int tid  = threadIdx.x;
    const int lane = tid & 31;
    const int wave = tid >> 5;
    const int b0   = blockIdx.x * ROWS;

    // ---- Phase 0: stage inputs as bf16 (perc rows 0..31, tech rows 32..63) ----
    for (int i = tid; i < TROWS * Dm; i += BLOCK) {
        const int r = i / Dm, c = i - r * Dm;
        float v = 0.f;
        if (r < ROWS) { if (b0 + r < B)        v = perc[(size_t)(b0 + r) * Dm + c]; }
        else          { if (b0 + r - ROWS < B) v = tech[(size_t)(b0 + r - ROWS) * Dm + c]; }
        sTmp[r * SSTR + c] = f2bf(v);
    }
    __syncthreads();

    // ---- proj_p / proj_t -> seq ----
    mm_tiles(sTmp,               SSTR, wsw + OFF_WP, bp, sSeq,               SSTR, false, 2, 12, Dm, wave, lane);
    mm_tiles(sTmp + ROWS * SSTR, SSTR, wsw + OFF_WT, bt, sSeq + ROWS * SSTR, SSTR, false, 2, 12, Dm, wave, lane);
    __syncthreads();

    // ---- QKV: [64 x 192] @ in_w^T -> [64 x 576] ----
    mm_tiles(sSeq, SSTR, wsw + OFF_IN, in_b, sQ, QSTR, false, 4, 36, Dm, wave, lane);
    __syncthreads();

    // ---- attention (2 tokens, 3 heads of 64); ctx overwrites Q region ----
    if (tid < ROWS * 3) {
        const int b = tid / 3, h = tid - 3 * b;
        const unsigned short* qp = sQ + (size_t)b * QSTR + h * 64;
        const unsigned short* qt = sQ + (size_t)(ROWS + b) * QSTR + h * 64;
        const unsigned short* kp = qp + Dm;
        const unsigned short* kt = qt + Dm;
        const unsigned short* vp = qp + 2 * Dm;
        const unsigned short* vt = qt + 2 * Dm;
        float s00 = 0.f, s01 = 0.f, s10 = 0.f, s11 = 0.f;
        for (int d = 0; d < 64; ++d) {
            const float a = bf2f(qp[d]), bq = bf2f(qt[d]);
            const float c = bf2f(kp[d]), e  = bf2f(kt[d]);
            s00 += a * c; s01 += a * e; s10 += bq * c; s11 += bq * e;
        }
        const float sc = 0.125f;   // 1/sqrt(64)
        s00 *= sc; s01 *= sc; s10 *= sc; s11 *= sc;
        const float m0 = fmaxf(s00, s01);
        const float e00 = __expf(s00 - m0), e01 = __expf(s01 - m0);
        const float r0 = 1.f / (e00 + e01), a00 = e00 * r0, a01 = e01 * r0;
        const float m1 = fmaxf(s10, s11);
        const float e10 = __expf(s10 - m1), e11 = __expf(s11 - m1);
        const float r1 = 1.f / (e10 + e11), a10 = e10 * r1, a11 = e11 * r1;
        unsigned short* cp = sQ + (size_t)b * QSTR + h * 64;
        unsigned short* ct = sQ + (size_t)(ROWS + b) * QSTR + h * 64;
        for (int d = 0; d < 64; ++d) {
            const float vpd = bf2f(vp[d]), vtd = bf2f(vt[d]);
            cp[d] = f2bf(a00 * vpd + a01 * vtd);
            ct[d] = f2bf(a10 * vpd + a11 * vtd);
        }
    }
    __syncthreads();

    // ---- out-proj: ctx @ out_w^T -> sTmp ----
    mm_tiles(sQ, QSTR, wsw + OFF_OUT, out_b, sTmp, SSTR, false, 4, 12, Dm, wave, lane);
    __syncthreads();

    // ---- x = LN1(seq + attn_out) in sSeq ----
    ln_residual(sSeq, sTmp, ln1_g, ln1_b, tid);
    __syncthreads();

    // ---- FFN1 (relu) -> [64 x 384] hidden (reuse sQ) ----
    mm_tiles(sSeq, SSTR, wsw + OFF_F1, ffn_b1, sQ, QSTR, true, 4, 24, Dm, wave, lane);
    __syncthreads();

    // ---- FFN2: hidden (K=384) @ ffn_w2^T -> sTmp ----
    mm_tiles(sQ, QSTR, wsw + OFF_F2, ffn_b2, sTmp, SSTR, false, 4, 12, 2 * Dm, wave, lane);
    __syncthreads();

    // ---- x = LN2(x + ffn) in sSeq ----
    ln_residual(sSeq, sTmp, ln2_g, ln2_b, tid);
    __syncthreads();

    // ---- z = mean over tokens -> sTmp rows 0..31 ----
    for (int i = tid; i < ROWS * Dm; i += BLOCK) {
        const int r = i / Dm, c = i - r * Dm;
        sTmp[r * SSTR + c] =
            f2bf(0.5f * (bf2f(sSeq[r * SSTR + c]) + bf2f(sSeq[(ROWS + r) * SSTR + c])));
    }
    __syncthreads();

    // ---- heads: cls hidden (relu) -> sSeq rows 0..31 ; fp -> sSeq rows 32..63 ----
    mm_tiles(sTmp, SSTR, wsw + OFF_CLS1, cls_b1, sSeq,               SSTR, true,  2, 12, Dm, wave, lane);
    mm_tiles(sTmp, SSTR, wsw + OFF_FP,   fp_b,   sSeq + ROWS * SSTR, SSTR, false, 2,  8, Dm, wave, lane);
    __syncthreads();

    // ---- logits -> sigmoid probs (3 per row) ----
    if (tid < ROWS * 3) {
        const int b = tid / 3, j = tid - 3 * b;
        float acc = cls_b2[j];
        for (int k = 0; k < Dm; ++k) acc += bf2f(sSeq[b * SSTR + k]) * cls_w2[j * Dm + k];
        const float pr = 1.0f / (1.0f + __expf(-acc));
        if (b0 + b < B) out[(size_t)j * B + b0 + b] = pr;
    }
    // ---- fingerprint L2-normalize + store ----
    if (tid < ROWS) {
        const int b = tid;
        const unsigned short* fpv = sSeq + (size_t)(ROWS + b) * SSTR;
        float ss = 0.f;
        for (int i = 0; i < 128; ++i) { const float v = bf2f(fpv[i]); ss += v * v; }
        const float scl = 1.0f / fmaxf(sqrtf(ss), 1e-12f);
        if (b0 + b < B) {
            float* dst = out + (size_t)3 * B + (size_t)(b0 + b) * 128;
            for (int i = 0; i < 128; ++i) dst[i] = bf2f(fpv[i]) * scl;
        }
    }
}

static void cvt(const float* src, unsigned short* dst, int n, hipStream_t s) {
    cvt_bf16_kernel<<<(n + 255) / 256, 256, 0, s>>>(src, dst, n);
}

extern "C" void kernel_launch(void* const* d_in, const int* in_sizes, int n_in,
                              void* d_out, int out_size, void* d_ws, size_t ws_size,
                              hipStream_t stream) {
    const float* perc   = (const float*)d_in[0];
    const float* tech   = (const float*)d_in[1];
    const float* Wp     = (const float*)d_in[2];
    const float* bp     = (const float*)d_in[3];
    const float* Wt     = (const float*)d_in[4];
    const float* bt     = (const float*)d_in[5];
    const float* in_w   = (const float*)d_in[6];
    const float* in_b   = (const float*)d_in[7];
    const float* out_w  = (const float*)d_in[8];
    const float* out_b  = (const float*)d_in[9];
    const float* ffn_w1 = (const float*)d_in[10];
    const float* ffn_b1 = (const float*)d_in[11];
    const float* ffn_w2 = (const float*)d_in[12];
    const float* ffn_b2 = (const float*)d_in[13];
    const float* ln1_g  = (const float*)d_in[14];
    const float* ln1_b  = (const float*)d_in[15];
    const float* ln2_g  = (const float*)d_in[16];
    const float* ln2_b  = (const float*)d_in[17];
    const float* cls_w1 = (const float*)d_in[18];
    const float* cls_b1 = (const float*)d_in[19];
    const float* cls_w2 = (const float*)d_in[20];
    const float* cls_b2 = (const float*)d_in[21];
    const float* fp_w   = (const float*)d_in[22];
    const float* fp_b   = (const float*)d_in[23];

    unsigned short* wsw = (unsigned short*)d_ws;

    // prepack weights to bf16 (row n stays K-contiguous -> B-fragment friendly)
    cvt(Wp,     wsw + OFF_WP,   Dm * Dm,     stream);
    cvt(Wt,     wsw + OFF_WT,   Dm * Dm,     stream);
    cvt(in_w,   wsw + OFF_IN,   3 * Dm * Dm, stream);
    cvt(out_w,  wsw + OFF_OUT,  Dm * Dm,     stream);
    cvt(ffn_w1, wsw + OFF_F1,   2 * Dm * Dm, stream);
    cvt(ffn_w2, wsw + OFF_F2,   2 * Dm * Dm, stream);
    cvt(cls_w1, wsw + OFF_CLS1, Dm * Dm,     stream);
    cvt(fp_w,   wsw + OFF_FP,   128 * Dm,    stream);

    const int B = in_sizes[0] / Dm;
    const int grid = (B + ROWS - 1) / ROWS;
    const size_t smem = ((size_t)TROWS * SSTR * 2) * 2 + (size_t)TROWS * QSTR * 2; // 125,952 B

    (void)hipFuncSetAttribute((const void*)fusion_head_kernel,
                              hipFuncAttributeMaxDynamicSharedMemorySize, (int)smem);

    fusion_head_kernel<<<grid, BLOCK, smem, stream>>>(
        perc, tech, wsw, bp, bt, in_b, out_b, ffn_b1, ffn_b2,
        ln1_g, ln1_b, ln2_g, ln2_b, cls_b1, cls_w2, cls_b2, fp_b,
        (float*)d_out, B);
}
